// TemporalAttention_82429012345096
// MI455X (gfx1250) — matrix operand order repaired and checked by reading, not compiled, not run on hardware
//
#include <hip/hip_runtime.h>
#include <hip/hip_bf16.h>

// ---------------------------------------------------------------------------
// TemporalAttention for MI455X (gfx1250, wave32, WMMA bf16 f32-accumulate)
//
// Stage 0: prep_kernel  — convert Wq/Wk/Wv fp32 -> bf16 (row-major), zero-fill
//                         the j=16..31 padding of v_t.
// Stage 1: proj_kernel  — 64 batched GEMMs W[256x256] @ X_n[256x1024] (+bias)
//                         for q,k,v simultaneously. bf16 WMMA 16x16x32.
// Stage 2: attn_kernel  — per (b,c): S = Q K^T / 16 (K-dim 1024 split over 4
//                         waves), softmax, O = P V, out = gamma*O + x.
// Scratch: q_s,k_s bf16 [64][256][1024]; v_t bf16 [4][256][1024][32] (padded);
//          wbf bf16 3x[256][256].  Total ~128.4 MB.
// ---------------------------------------------------------------------------

typedef __attribute__((ext_vector_type(16))) __bf16 v16bf;
typedef __attribute__((ext_vector_type(2)))  __bf16 v2bf;
typedef __attribute__((ext_vector_type(8)))  float  v8f;

union BF16x16 {
    v16bf        v;
    unsigned int u32[8];
};

// Packed f32->bf16 conversion: native v_cvt_pk_bf16_f32 when available.
__device__ __forceinline__ unsigned int pack2bf(float lo, float hi) {
#if __has_builtin(__builtin_amdgcn_cvt_pk_bf16_f32)
    v2bf r = __builtin_amdgcn_cvt_pk_bf16_f32(lo, hi);
    return __builtin_bit_cast(unsigned int, r);
#else
    union { v2bf v; unsigned int u; } u;
    u.v[0] = (__bf16)lo;
    u.v[1] = (__bf16)hi;
    return u.u;
#endif
}
__device__ __forceinline__ unsigned short f2bf(float f) {
#if __has_builtin(__builtin_amdgcn_cvt_pk_bf16_f32)
    v2bf r = __builtin_amdgcn_cvt_pk_bf16_f32(f, 0.f);
    union { v2bf v; unsigned short u[2]; } u;
    u.v = r;
    return u.u[0];
#else
    union { __bf16 b; unsigned short u; } u;
    u.b = (__bf16)f;
    return u.u;
#endif
}

#define CDIM   256
#define PTOT   1024
#define XT_STR 132   // dword stride of LDS X-tile rows: 16B-aligned, bank Δ=4

// ---------------------------------------------------------------------------
// Stage 0: W fp32->bf16 once; zero v_t's j=16..31 padding.
// ---------------------------------------------------------------------------
__global__ void __launch_bounds__(256)
prep_kernel(const float* __restrict__ wq, const float* __restrict__ wk,
            const float* __restrict__ wv,
            unsigned int* __restrict__ wbf32,   // 3 * 32768 packed pairs
            unsigned int* __restrict__ vt32)    // v_t viewed as u32
{
    const float* wmat[3] = { wq, wk, wv };
    const int tid    = blockIdx.x * blockDim.x + threadIdx.x;
    const int stride = gridDim.x * blockDim.x;

    for (int i = tid; i < 3 * 32768; i += stride) {
        int pr = i >> 15, off = i & 32767;
        const float* w = wmat[pr];
        wbf32[i] = pack2bf(w[2 * off], w[2 * off + 1]);
    }
    // padding: per (b,c,p) the u32 slots [8,16) of the 16-u32 j-row
    for (int i = tid; i < 4 * 256 * 1024 * 8; i += stride) {
        int bcp = i >> 3, jz = i & 7;
        vt32[bcp * 16 + 8 + jz] = 0u;
    }
}

// ---------------------------------------------------------------------------
// Stage 1: fused q/k/v projection.  grid = (16 p-blocks, 64 n), block = 512.
// Each wave: one 16-wide d-tile, 4 p-subtiles of 16, 3 projections.
// ---------------------------------------------------------------------------
__global__ void __launch_bounds__(512)
proj_kernel(const float* __restrict__ x,
            const unsigned short* __restrict__ wbf,   // 3 x [256][256] bf16
            const float* __restrict__ bq,
            const float* __restrict__ bk,
            const float* __restrict__ bv,
            unsigned short* __restrict__ q_s,
            unsigned short* __restrict__ k_s,
            unsigned short* __restrict__ v_t)
{
    // X tile transposed + bf16-packed: xT[p_local][c/2], padded stride 132
    __shared__ unsigned int xT[64 * XT_STR];

    const int tid = threadIdx.x;
    const int n   = blockIdx.y;        // 0..63  (= b*16 + j)
    const int p0  = blockIdx.x * 64;   // p block base

    // ---- stage X[256c x 64p] fp32 -> LDS bf16 pairs (transposed) ----------
    {
        const int p   = tid & 63;
        const int cp0 = tid >> 6;                       // 0..7
        const float* xn = x + (size_t)n * (CDIM * PTOT) + p0 + p;
        #pragma unroll
        for (int i = 0; i < 16; ++i) {
            int   cp = cp0 + (i << 3);                  // 0..127
            int   c  = cp << 1;
            float a  = xn[(size_t)c * PTOT];
            float bb = xn[(size_t)(c + 1) * PTOT];
            xT[p * XT_STR + cp] = pack2bf(a, bb);
        }
    }
    __syncthreads();

    const int lane  = tid & 31;
    const int wave  = tid >> 5;        // 0..15
    const int m     = lane & 15;
    const int hh    = lane >> 4;       // half-wave select
    const int dtile = wave * 16;

    v8f zero = {};
    v8f acc[3][4];
    #pragma unroll
    for (int pr = 0; pr < 3; ++pr)
        #pragma unroll
        for (int ps = 0; ps < 4; ++ps) acc[pr][ps] = zero;

    for (int cc = 0; cc < CDIM; cc += 32) {
        // A fragments (bf16 weights), ISA 16-bit A layout:
        // lanes<16: V0-3 K=cc+0..7, V4-7 K=cc+16..23; lanes>=16: +8
        BF16x16 afrag[3];
        #pragma unroll
        for (int pr = 0; pr < 3; ++pr) {
            const unsigned short* wrow = wbf + pr * 65536 + (dtile + m) * CDIM;
            *(uint4*)&afrag[pr].u32[0] = *(const uint4*)(wrow + cc + hh * 8);
            *(uint4*)&afrag[pr].u32[4] = *(const uint4*)(wrow + cc + 16 + hh * 8);
        }
        // B fragments from LDS, ISA 16-bit B layout:
        // VGPR v<4: k=cc+8h+2v,+1; v>=4: k=cc+16+8h+2(v-4),+1 (h = lane half)
        BF16x16 bfrag[4];
        #pragma unroll
        for (int ps = 0; ps < 4; ++ps) {
            int prow = (ps * 16 + m) * XT_STR;
            int cpb  = (cc >> 1) + hh * 4;
            *(uint4*)&bfrag[ps].u32[0] = *(const uint4*)&xT[prow + cpb];
            *(uint4*)&bfrag[ps].u32[4] = *(const uint4*)&xT[prow + cpb + 8];
        }
        #pragma unroll
        for (int pr = 0; pr < 3; ++pr)
            #pragma unroll
            for (int ps = 0; ps < 4; ++ps)
                acc[pr][ps] = __builtin_amdgcn_wmma_f32_16x16x32_bf16(
                    false, afrag[pr].v, false, bfrag[ps].v,
                    (short)0, acc[pr][ps], false, false);
    }

    // ---- bias + store (q_s/k_s row-major bf16, v -> [b][c][p][32j]) ------
    const float* bias[3] = { bq, bk, bv };
    const int b4 = n >> 4, jj = n & 15;
    #pragma unroll
    for (int pr = 0; pr < 3; ++pr) {
        float bl[8];
        #pragma unroll
        for (int r = 0; r < 8; ++r) bl[r] = bias[pr][dtile + hh * 8 + r];
        #pragma unroll
        for (int ps = 0; ps < 4; ++ps) {
            #pragma unroll
            for (int r = 0; r < 8; ++r) {
                float val = acc[pr][ps][r] + bl[r];
                int d = dtile + hh * 8 + r;          // C/D layout: row = r (+8 hi half)
                int p = p0 + ps * 16 + m;            // col = lane&15
                if (pr == 0)
                    q_s[((size_t)n << 18) + (size_t)d * PTOT + p] = f2bf(val);
                else if (pr == 1)
                    k_s[((size_t)n << 18) + (size_t)d * PTOT + p] = f2bf(val);
                else
                    v_t[((((size_t)b4 * CDIM + d) << 10) + p) * 32 + jj] = f2bf(val);
            }
        }
    }
}

// ---------------------------------------------------------------------------
// Stage 2: attention per (b,c).  grid = 1024, block = 128 (4 waves).
// ---------------------------------------------------------------------------
__global__ void __launch_bounds__(128)
attn_kernel(const unsigned short* __restrict__ q_s,
            const unsigned short* __restrict__ k_s,
            const unsigned short* __restrict__ v_t,
            const float* __restrict__ x,
            const float* __restrict__ gamma,
            float* __restrict__ out)
{
    __shared__ float                         sRed[4][256];
    __shared__ __align__(16) unsigned short  pSc[4][256];

    const int tid  = threadIdx.x;
    const int lane = tid & 31;
    const int wave = tid >> 5;
    const int m    = lane & 15;
    const int hh   = lane >> 4;
    const int bc   = blockIdx.x;
    const int b    = bc >> 8;
    const int c    = bc & 255;

    // Row m of Q / K for this (b,c): contiguous 1024 bf16
    const unsigned short* qrow = q_s + (((size_t)(b * 16 + m)) << 18) + (size_t)c * 1024;
    const unsigned short* krow = k_s + (((size_t)(b * 16 + m)) << 18) + (size_t)c * 1024;

    // ---- S partial = Q K^T over this wave's 256-wide p slice --------------
    v8f sAcc = {};
    for (int t = 0; t < 8; ++t) {
        int kc = wave * 256 + t * 32;
        BF16x16 aq, bkf;
        // A: Q row contiguous -> two b128 loads per lane
        *(uint4*)&aq.u32[0] = *(const uint4*)(qrow + kc + hh * 8);
        *(uint4*)&aq.u32[4] = *(const uint4*)(qrow + kc + 16 + hh * 8);
        // B: K row j=m, same k order as A -> two b128 loads
        *(uint4*)&bkf.u32[0] = *(const uint4*)(krow + kc + hh * 8);
        *(uint4*)&bkf.u32[4] = *(const uint4*)(krow + kc + 16 + hh * 8);
        sAcc = __builtin_amdgcn_wmma_f32_16x16x32_bf16(
            false, aq.v, false, bkf.v, (short)0, sAcc, false, false);
    }

    // ---- cross-wave reduce of S ------------------------------------------
    #pragma unroll
    for (int r = 0; r < 8; ++r)
        sRed[wave][(hh * 8 + r) * 16 + m] = sAcc[r];
    __syncthreads();

    float s[8];
    #pragma unroll
    for (int r = 0; r < 8; ++r) {
        float a = 0.f;
        #pragma unroll
        for (int w4 = 0; w4 < 4; ++w4) a += sRed[w4][(hh * 8 + r) * 16 + m];
        s[r] = a * 0.0625f;                       // 1/sqrt(256)
    }

    // ---- softmax over j (16 lanes of each half-wave hold one row) --------
    float pv[8];
    #pragma unroll
    for (int r = 0; r < 8; ++r) {
        float v  = s[r];
        float mx = v;
        #pragma unroll
        for (int d = 1; d < 16; d <<= 1) mx = fmaxf(mx, __shfl_xor(mx, d, 32));
        float e  = __expf(v - mx);
        float sm = e;
        #pragma unroll
        for (int d = 1; d < 16; d <<= 1) sm += __shfl_xor(sm, d, 32);
        pv[r] = e / sm;
    }

    // ---- restage P through LDS into A-layout (K padded 16 -> 32) ---------
    #pragma unroll
    for (int r = 0; r < 8; ++r)
        pSc[wave][(hh * 8 + r) * 16 + m] = f2bf(pv[r]);
    __syncthreads();

    BF16x16 pa;
    *(uint4*)&pa.u32[0] = *(const uint4*)&pSc[wave][m * 16 + hh * 8];  // 8 halves
    #pragma unroll
    for (int v = 4; v < 8; ++v) pa.u32[v] = 0u;   // K = 16..31 zero

    // ---- O = P V, fused epilogue out = gamma*O + x -----------------------
    const float  g      = gamma[0];
    const size_t vtbase = ((size_t)(b * CDIM + c)) << 15;   // *1024*32
    for (int t = 0; t < 16; ++t) {
        int p0 = (wave * 16 + t) * 16;
        const unsigned short* vrow = v_t + vtbase + (size_t)(p0 + m) * 32 + hh * 8;
        BF16x16 vb;          // j = hh*8 + 0..7, then 16 + hh*8 + 0..7; j>=16 is zero padding
        *(uint4*)&vb.u32[0] = *(const uint4*)(vrow);
        *(uint4*)&vb.u32[4] = *(const uint4*)(vrow + 16);
        v8f o = {};
        o = __builtin_amdgcn_wmma_f32_16x16x32_bf16(
            false, pa.v, false, vb.v, (short)0, o, false, false);
        #pragma unroll
        for (int r = 0; r < 8; ++r) {
            int    i   = hh * 8 + r;
            size_t idx = ((size_t)b << 22) + ((size_t)c << 14) + ((size_t)i << 10) + p0 + m;
            out[idx] = g * o[r] + x[idx];
        }
    }
}

// ---------------------------------------------------------------------------
extern "C" void kernel_launch(void* const* d_in, const int* in_sizes, int n_in,
                              void* d_out, int out_size, void* d_ws, size_t ws_size,
                              hipStream_t stream) {
    const float* x  = (const float*)d_in[0];
    const float* wq = (const float*)d_in[1];
    const float* bq = (const float*)d_in[2];
    const float* wk = (const float*)d_in[3];
    const float* bk = (const float*)d_in[4];
    const float* wv = (const float*)d_in[5];
    const float* bv = (const float*)d_in[6];
    const float* gm = (const float*)d_in[7];

    unsigned short* q_s = (unsigned short*)d_ws;              // 2^24 bf16 (32 MB)
    unsigned short* k_s = q_s + ((size_t)1 << 24);            // 2^24 bf16 (32 MB)
    unsigned short* v_t = k_s + ((size_t)1 << 24);            // 2^25 bf16 (64 MB, padded)
    unsigned short* wbf = v_t + ((size_t)1 << 25);            // 3*65536 bf16 (384 KB)

    prep_kernel<<<4096, 256, 0, stream>>>(wq, wk, wv,
                                          (unsigned int*)wbf,
                                          (unsigned int*)v_t);
    dim3 g1(16, 64);   // (p-blocks, n)
    proj_kernel<<<g1, 512, 0, stream>>>(x, wbf, bq, bk, bv, q_s, k_s, v_t);
    attn_kernel<<<1024, 128, 0, stream>>>(q_s, k_s, v_t, x, gm, (float*)d_out);
}
